// GraphAttentionLayer_103079215161
// MI455X (gfx1250) — matrix operand order repaired and checked by reading, not compiled, not run on hardware
//
#include <hip/hip_runtime.h>
#include <hip/hip_bf16.h>

typedef __attribute__((ext_vector_type(16))) _Float16     v16h;
typedef __attribute__((ext_vector_type(2)))  _Float16     v2h;
typedef __attribute__((ext_vector_type(8)))  float        v8f;
typedef __attribute__((ext_vector_type(2)))  float        v2f;
typedef __attribute__((ext_vector_type(8)))  unsigned int v8u;

#define N_NODES   4096
#define IN_FEAT   256
#define N_HEADS   8
#define N_HID     32
#define OUT_COLS  256   // N_HEADS * N_HID

// ---------------------------------------------------------------------------
// Kernel 1: g = h @ W (fp32, V_WMMA_F32_16X16X4_F32). One wave per 16x16
// tile, K streamed in steps of 4. Emits g (fp32, for score dots) and a
// transposed f16 copy gT[col][node] for the attention kernel's B-loads.
// ---------------------------------------------------------------------------
__global__ __launch_bounds__(256) void gat_gemm_hw(
    const float* __restrict__ h, const float* __restrict__ W,
    float* __restrict__ g, _Float16* __restrict__ gT) {
  const int wave = threadIdx.x >> 5;
  const int lane = threadIdx.x & 31;
  const int tile = blockIdx.x * 8 + wave;   // 4096 tiles total
  const int i0 = (tile >> 4) * 16;          // row block
  const int j0 = (tile & 15) * 16;          // col block
  const int half = lane >> 4;
  const int m = lane & 15;

  v8f acc = {};
  for (int k0 = 0; k0 < IN_FEAT; k0 += 4) {
    const int ka = k0 + 2 * half;
    v2f a, b;
    a.x = h[(i0 + m) * IN_FEAT + ka];
    a.y = h[(i0 + m) * IN_FEAT + ka + 1];
    b.x = W[ka * OUT_COLS + j0 + m];
    b.y = W[(ka + 1) * OUT_COLS + j0 + m];
    acc = __builtin_amdgcn_wmma_f32_16x16x4_f32(
        false, a, false, b, (short)0, acc, false, false);
  }
#pragma unroll
  for (int r = 0; r < 8; ++r) {
    const int M = i0 + r + 8 * half;
    const int N = j0 + m;
    g[M * OUT_COLS + N] = acc[r];
    gT[N * N_NODES + M] = (_Float16)acc[r];
  }
}

// ---------------------------------------------------------------------------
// Kernel 2: per (node,head):
//   s_src[n][h]  = g[n,h,:].a[:32]
//   E1[h][n]     = (f16) exp(s_dst)        (pos-branch column factor)
//   E2[h][n]     = (f16) exp(0.2*s_dst)    (neg-branch column factor)
// Split half-tables keep the attention loop's loads contiguous & pk-ready.
// ---------------------------------------------------------------------------
__global__ __launch_bounds__(256) void gat_scores(
    const float* __restrict__ g, const float* __restrict__ a,
    float* __restrict__ s_src, _Float16* __restrict__ E1t,
    _Float16* __restrict__ E2t) {
  const int t = blockIdx.x * 256 + threadIdx.x;   // 32768 = 4096*8
  const int n = t >> 3;
  const int hd = t & 7;
  const float* gp = g + n * OUT_COLS + hd * N_HID;
  float ss = 0.f, sd = 0.f;
#pragma unroll
  for (int f = 0; f < N_HID; ++f) {
    const float v = gp[f];
    ss += v * a[f];
    sd += v * a[N_HID + f];
  }
  s_src[t] = ss;
  E1t[hd * N_NODES + n] = (_Float16)__expf(sd);
  E2t[hd * N_NODES + n] = (_Float16)__expf(0.2f * sd);
}

// ---------------------------------------------------------------------------
// Kernel 3: fused masked-softmax attention + aggregation.
// Block (8 waves) per 16-row block; wave = head; all waves share the same
// adjacency stream through WGP$ (HBM reads adjacency exactly once, 64 MB).
// Per 32-column chunk per lane: 12 b128 loads total; A-tile built with
// packed f16 math using exp(leaky_relu(x)) = max(exp(x), exp(0.2x)):
//   per element-pair: 2x v_pk_mul_f16 + v_pk_max_num_f16 + 3 int ops (mask)
// then 3x v_wmma_f32_16x16x32_f16 (acc0, acc1, rowsums-vs-ones).
// ---------------------------------------------------------------------------
__global__ __launch_bounds__(256) void gat_attention(
    const int* __restrict__ adj, const float* __restrict__ s_src,
    const _Float16* __restrict__ E1t, const _Float16* __restrict__ E2t,
    const _Float16* __restrict__ gT, float* __restrict__ out) {
  const int lane = threadIdx.x & 31;
  const int hd = threadIdx.x >> 5;          // head
  const int i0 = blockIdx.x * 16;
  const int half = lane >> 4;
  const int m = lane & 15;

  const float ss = s_src[(i0 + m) * N_HEADS + hd];
  const _Float16 e1h = (_Float16)__expf(ss);
  const _Float16 e2h = (_Float16)__expf(0.2f * ss);
  const v2h e1pp = {e1h, e1h};
  const v2h e2pp = {e2h, e2h};

  const int* ap = adj + (i0 + m) * N_NODES;
  // half-pair (uint) views of the column-factor tables for this head
  const unsigned int* e1p = (const unsigned int*)(E1t + hd * N_NODES);
  const unsigned int* e2p = (const unsigned int*)(E2t + hd * N_NODES);
  const _Float16* b0p = gT + (hd * N_HID + m) * N_NODES;
  const _Float16* b1p = gT + (hd * N_HID + 16 + m) * N_NODES;

  v16h ones;
#pragma unroll
  for (int e = 0; e < 16; ++e) ones[e] = (_Float16)1.0f;

  v8f acc0 = {}, acc1 = {}, accr = {};

  for (int j0 = 0; j0 < N_NODES; j0 += 32) {
    const int r0 = 8 * half;              // A-layout K-run 0 (elements 0..7)
    const int r1 = 16 + 8 * half;         // A-layout K-run 1 (elements 8..15)
    const uint4 q0 = *(const uint4*)(ap + j0 + r0);
    const uint4 q1 = *(const uint4*)(ap + j0 + r0 + 4);
    const uint4 q2 = *(const uint4*)(ap + j0 + r1);
    const uint4 q3 = *(const uint4*)(ap + j0 + r1 + 4);
    const int jp = j0 >> 1;               // half-pair index base
    const uint4 f1a = *(const uint4*)(e1p + jp + 4 * half);      // run0 E1
    const uint4 f1b = *(const uint4*)(e1p + jp + 8 + 4 * half);  // run1 E1
    const uint4 f2a = *(const uint4*)(e2p + jp + 4 * half);      // run0 E2
    const uint4 f2b = *(const uint4*)(e2p + jp + 8 + 4 * half);  // run1 E2
    __builtin_prefetch(ap + j0 + 32, 0, 1);   // speculative: next adj chunk

    const unsigned int qa[16] = {q0.x, q0.y, q0.z, q0.w, q1.x, q1.y, q1.z, q1.w,
                                 q2.x, q2.y, q2.z, q2.w, q3.x, q3.y, q3.z, q3.w};
    const unsigned int f1d[8] = {f1a.x, f1a.y, f1a.z, f1a.w,
                                 f1b.x, f1b.y, f1b.z, f1b.w};
    const unsigned int f2d[8] = {f2a.x, f2a.y, f2a.z, f2a.w,
                                 f2b.x, f2b.y, f2b.z, f2b.w};

    v8u wu;
#pragma unroll
    for (int p = 0; p < 8; ++p) {                   // pair of A elements
      const v2h c1 = __builtin_bit_cast(v2h, f1d[p]);
      const v2h c2 = __builtin_bit_cast(v2h, f2d[p]);
      const v2h t0 = e1pp * c1;                     // v_pk_mul_f16
      const v2h t1 = e2pp * c2;                     // v_pk_mul_f16
      const v2h tm = __builtin_elementwise_max(t0, t1);  // v_pk_max_num_f16
      const unsigned int qp  = qa[2 * p] | (qa[2 * p + 1] << 16);
      const unsigned int msk = qp * 0xFFFFu;        // {0,1}->{0x0000,0xFFFF}
      wu[p] = __builtin_bit_cast(unsigned int, tm) & msk;
    }
    const v16h A = __builtin_bit_cast(v16h, wu);

    typedef __attribute__((ext_vector_type(8))) _Float16 v8h;
    const v16h B0 = __builtin_shufflevector(*(const v8h*)(b0p + j0 + r0), *(const v8h*)(b0p + j0 + r1), 0, 1, 2, 3, 4, 5, 6, 7, 8, 9, 10, 11, 12, 13, 14, 15);
    const v16h B1 = __builtin_shufflevector(*(const v8h*)(b1p + j0 + r0), *(const v8h*)(b1p + j0 + r1), 0, 1, 2, 3, 4, 5, 6, 7, 8, 9, 10, 11, 12, 13, 14, 15);
    acc0 = __builtin_amdgcn_wmma_f32_16x16x32_f16(
        false, A, false, B0, (short)0, acc0, false, false);
    acc1 = __builtin_amdgcn_wmma_f32_16x16x32_f16(
        false, A, false, B1, (short)0, acc1, false, false);
    accr = __builtin_amdgcn_wmma_f32_16x16x32_f16(
        false, A, false, ones, (short)0, accr, false, false);
  }

  // accr[r] holds rowsum for M = r + 8*half in this very lane (D-layout):
#pragma unroll
  for (int r = 0; r < 8; ++r) {
    const int M = i0 + r + 8 * half;
    const float inv = (accr[r] != 0.f) ? (1.f / accr[r]) : 0.f;
    out[M * OUT_COLS + hd * N_HID + m]      = acc0[r] * inv;
    out[M * OUT_COLS + hd * N_HID + 16 + m] = acc1[r] * inv;
  }
}

// ---------------------------------------------------------------------------
extern "C" void kernel_launch(void* const* d_in, const int* in_sizes, int n_in,
                              void* d_out, int out_size, void* d_ws, size_t ws_size,
                              hipStream_t stream) {
  const float* h   = (const float*)d_in[0];   // 4096 x 256
  const int*   adj = (const int*)d_in[1];     // 4096 x 4096 (x1)
  const float* W   = (const float*)d_in[2];   // 256 x 256
  const float* a   = (const float*)d_in[3];   // 64
  float* out = (float*)d_out;                 // 4096 x 256

  char* ws = (char*)d_ws;
  float*    g     = (float*)ws;                                   // 4 MB
  _Float16* gT    = (_Float16*)(ws + (4u << 20));                 // 2 MB
  float*    s_src = (float*)(ws + (6u << 20));                    // 128 KB
  _Float16* E1t   = (_Float16*)(ws + (6u << 20) + (128u << 10));  // 64 KB
  _Float16* E2t   = (_Float16*)(ws + (6u << 20) + (192u << 10));  // 64 KB

  gat_gemm_hw<<<512, 256, 0, stream>>>(h, W, g, gT);
  gat_scores<<<128, 256, 0, stream>>>(g, a, s_src, E1t, E2t);
  gat_attention<<<256, 256, 0, stream>>>(adj, s_src, E1t, E2t, gT, out);
}
